// NaiveMatryoshkaLinear_59820304498925
// MI455X (gfx1250) — matrix operand order repaired and checked by reading, not compiled, not run on hardware
//
#include <hip/hip_runtime.h>
#include <stdint.h>

typedef __attribute__((ext_vector_type(16))) __bf16    v16bf;
typedef __attribute__((ext_vector_type(8)))  float     v8f;
typedef __attribute__((ext_vector_type(4)))  float     v4f;
typedef __attribute__((ext_vector_type(4)))  unsigned  v4u;
typedef __attribute__((ext_vector_type(2)))  unsigned  v2u;
typedef __attribute__((ext_vector_type(4)))  int       v4i;

#if __has_builtin(__builtin_amdgcn_global_load_async_to_lds_b128) && \
    __has_builtin(__builtin_amdgcn_s_wait_asynccnt)
#define HAS_ASYNC_LDS 1
#else
#define HAS_ASYNC_LDS 0
#endif

#define BM 64
#define BN 128
#define BK 32
#define LDSROW 20                           /* uints/row: 16 data (32 bf16) + 4 pad */
#define A_ELEMS (BM * LDSROW)               /* 1280 uints */
#define B_ELEMS (BN * LDSROW)               /* 2560 uints */
#define STAGE_U (2 * A_ELEMS + 2 * B_ELEMS) /* 7680 uints = 30 KB/stage */
#define AHI_OFF 0
#define ALO_OFF (A_ELEMS)
#define BHI_OFF (2 * A_ELEMS)
#define BLO_OFF (2 * A_ELEMS + B_ELEMS)

// ---------------- bf16 split helpers (used only in pre-pass / fallback) -----
static __device__ __forceinline__ unsigned bf16_rne(float f) {
  unsigned u = __float_as_uint(f);
  u += 0x7FFFu + ((u >> 16) & 1u);
  return u >> 16;
}
static __device__ __forceinline__ float bf16_val(unsigned h) {
  return __uint_as_float(h << 16);
}

// ---------------- address-space-typed pointers for async copies -------------
typedef __attribute__((address_space(1))) v4i gbl_v4i;  // global (AS1)
typedef __attribute__((address_space(3))) v4i lds_v4i;  // LDS (AS3)

static __device__ __forceinline__ void async_cp16(const void* g, void* l) {
#if HAS_ASYNC_LDS
  // generic LDS address keeps the LDS byte offset in the low 32 bits
  __builtin_amdgcn_global_load_async_to_lds_b128(
      (gbl_v4i*)(uintptr_t)g,
      (lds_v4i*)(unsigned)(uintptr_t)l,
      0, 0);
#endif
}

// ---------------- WMMA fragment loads from LDS ------------------------------
// A-operand 16x32 bf16: lane-half 0 holds K{0..7,16..23}, half 1 K{8..15,24..31}
static __device__ __forceinline__ v16bf load_frag_a(const unsigned* base, int row, int half) {
  union { v4u q[2]; v16bf v; } f;
  const unsigned* p = base + row * LDSROW + half * 4;
  f.q[0] = *(const v4u*)p;
  f.q[1] = *(const v4u*)(p + 8);
  return f.v;
}
// B-operand 32x16 bf16: lane-half 0 holds K{0..7,16..23}, half 1 K{8..15,24..31}
static __device__ __forceinline__ v16bf load_frag_b(const unsigned* base, int row, int half) {
  union { v4u q[2]; v16bf v; } f;
  const unsigned* p = base + row * LDSROW + half * 4;
  f.q[0] = *(const v4u*)p;
  f.q[1] = *(const v4u*)(p + 8);
  return f.v;
}

// ---------------- pre-pass: f32 -> (hi bf16, lo bf16) pair arrays -----------
__global__ __launch_bounds__(256)
void convert_split_kernel(const float* __restrict__ src,
                          unsigned* __restrict__ dhi,   // 2 bf16 per uint
                          unsigned* __restrict__ dlo,
                          int n4) {
  int idx    = blockIdx.x * blockDim.x + threadIdx.x;
  int stride = gridDim.x * blockDim.x;
  for (int i = idx; i < n4; i += stride) {
    v4f v = *(const v4f*)(src + (size_t)i * 4);
    unsigned h[4], l[4];
#pragma unroll
    for (int j = 0; j < 4; ++j) {
      float f = v[j];
      unsigned hh = bf16_rne(f);
      h[j] = hh;
      l[j] = bf16_rne(f - bf16_val(hh));
    }
    v2u hp = { h[0] | (h[1] << 16), h[2] | (h[3] << 16) };
    v2u lp = { l[0] | (l[1] << 16), l[2] | (l[3] << 16) };
    *(v2u*)(dhi + (size_t)i * 2) = hp;
    *(v2u*)(dlo + (size_t)i * 2) = lp;
  }
}

// ---------------- main GEMM -------------------------------------------------
template <bool PRE>
__global__ __launch_bounds__(256)
void matryoshka_wmma_kernel(const float* __restrict__ x,
                            const float* __restrict__ W0,
                            const float* __restrict__ W1,
                            const float* __restrict__ W2,
                            const unsigned* __restrict__ xhiU,
                            const unsigned* __restrict__ xloU,
                            const unsigned* __restrict__ whiU,
                            const unsigned* __restrict__ wloU,
                            float* __restrict__ out) {
  __shared__ unsigned lds[2 * STAGE_U];   // 60 KB double buffered

  const int tid  = threadIdx.x;
  const int lane = tid & 31;
  const int wave = tid >> 5;
  const int wm = wave >> 2;   // 0..1
  const int wn = wave & 3;    // 0..3
  const int ln = lane & 15;
  const int lh = lane >> 4;

  const int bx = blockIdx.x;
  const float* W; int K, outBase, nb; size_t segU;
  if (bx < 8)       { W = W0; K = 1024; outBase = 0;    nb = bx;      segU = 0;        }
  else if (bx < 16) { W = W1; K = 2048; outBase = 1024; nb = bx - 8;  segU = 524288;   }
  else              { W = W2; K = 4096; outBase = 2048; nb = bx - 16; segU = 1572864;  }

  const int rowBase  = blockIdx.y * BM;
  const int wRowBase = nb * BN;

  // ---- PRE path staging (bf16 bytes only) ----
  v4u aH, aL, bH[2], bL[2];               // sync fallback staging regs
  auto loadStagePre = [&](int k0) {
    int rA = tid >> 2, cA = tid & 3;      // 64 rows x 4 chunks of 16B
    size_t ga = (size_t)(rowBase + rA) * 2048 + (k0 >> 1) + cA * 4;
    aH = *(const v4u*)(xhiU + ga);
    aL = *(const v4u*)(xloU + ga);
#pragma unroll
    for (int i = 0; i < 2; ++i) {
      int c = tid + 256 * i;
      int rB = c >> 2, cB = c & 3;        // 128 rows x 4 chunks
      size_t gb = segU + (size_t)(wRowBase + rB) * (K >> 1) + (k0 >> 1) + cB * 4;
      bH[i] = *(const v4u*)(whiU + gb);
      bL[i] = *(const v4u*)(wloU + gb);
    }
  };
  auto storeStagePre = [&](unsigned* st) {
    int rA = tid >> 2, cA = tid & 3;
    *(v4u*)(st + AHI_OFF + rA * LDSROW + cA * 4) = aH;
    *(v4u*)(st + ALO_OFF + rA * LDSROW + cA * 4) = aL;
#pragma unroll
    for (int i = 0; i < 2; ++i) {
      int c = tid + 256 * i;
      int rB = c >> 2, cB = c & 3;
      *(v4u*)(st + BHI_OFF + rB * LDSROW + cB * 4) = bH[i];
      *(v4u*)(st + BLO_OFF + rB * LDSROW + cB * 4) = bL[i];
    }
  };
#if HAS_ASYNC_LDS
  auto stageAsync = [&](int k0, unsigned* st) {
    int rA = tid >> 2, cA = tid & 3;
    size_t ga = (size_t)(rowBase + rA) * 2048 + (k0 >> 1) + cA * 4;
    async_cp16(xhiU + ga, st + AHI_OFF + rA * LDSROW + cA * 4);
    async_cp16(xloU + ga, st + ALO_OFF + rA * LDSROW + cA * 4);
#pragma unroll
    for (int i = 0; i < 2; ++i) {
      int c = tid + 256 * i;
      int rB = c >> 2, cB = c & 3;
      size_t gb = segU + (size_t)(wRowBase + rB) * (K >> 1) + (k0 >> 1) + cB * 4;
      async_cp16(whiU + gb, st + BHI_OFF + rB * LDSROW + cB * 4);
      async_cp16(wloU + gb, st + BLO_OFF + rB * LDSROW + cB * 4);
    }
  };
#endif

  // ---- fallback path staging (convert f32 -> hi/lo bf16 in kernel) ----
  v4f aReg[2], bReg[4];
  auto loadStageF32 = [&](int k0) {
#pragma unroll
    for (int i = 0; i < 2; ++i) {
      int c = tid + 256 * i;
      int r = c >> 3, kc = (c & 7) << 2;
      aReg[i] = *(const v4f*)(x + (size_t)(rowBase + r) * 4096 + (k0 + kc));
    }
#pragma unroll
    for (int i = 0; i < 4; ++i) {
      int c = tid + 256 * i;
      int r = c >> 3, kc = (c & 7) << 2;
      bReg[i] = *(const v4f*)(W + (size_t)(wRowBase + r) * K + (k0 + kc));
    }
  };
  auto cvtStoreF32 = [&](unsigned* st) {
#pragma unroll
    for (int i = 0; i < 2; ++i) {
      int c = tid + 256 * i;
      int r = c >> 3, kc = (c & 7) << 2;
      unsigned h[4], l[4];
#pragma unroll
      for (int j = 0; j < 4; ++j) {
        float f = aReg[i][j];
        unsigned hh = bf16_rne(f);
        h[j] = hh; l[j] = bf16_rne(f - bf16_val(hh));
      }
      v2u hp = { h[0] | (h[1] << 16), h[2] | (h[3] << 16) };
      v2u lp = { l[0] | (l[1] << 16), l[2] | (l[3] << 16) };
      *(v2u*)(st + AHI_OFF + r * LDSROW + (kc >> 1)) = hp;
      *(v2u*)(st + ALO_OFF + r * LDSROW + (kc >> 1)) = lp;
    }
#pragma unroll
    for (int i = 0; i < 4; ++i) {
      int c = tid + 256 * i;
      int r = c >> 3, kc = (c & 7) << 2;
      unsigned h[4], l[4];
#pragma unroll
      for (int j = 0; j < 4; ++j) {
        float f = bReg[i][j];
        unsigned hh = bf16_rne(f);
        h[j] = hh; l[j] = bf16_rne(f - bf16_val(hh));
      }
      v2u hp = { h[0] | (h[1] << 16), h[2] | (h[3] << 16) };
      v2u lp = { l[0] | (l[1] << 16), l[2] | (l[3] << 16) };
      *(v2u*)(st + BHI_OFF + r * LDSROW + (kc >> 1)) = hp;
      *(v2u*)(st + BLO_OFF + r * LDSROW + (kc >> 1)) = lp;
    }
  };

  v8f acc[2][2] = {};

  auto computeStep = [&](const unsigned* cur) {
    v16bf ah[2], al[2], bh[2], bl[2];
#pragma unroll
    for (int t = 0; t < 2; ++t) {
      int r = wm * 32 + t * 16 + ln;
      ah[t] = load_frag_a(cur + AHI_OFF, r, lh);
      al[t] = load_frag_a(cur + ALO_OFF, r, lh);
    }
#pragma unroll
    for (int u = 0; u < 2; ++u) {
      int r = wn * 32 + u * 16 + ln;
      bh[u] = load_frag_b(cur + BHI_OFF, r, lh);
      bl[u] = load_frag_b(cur + BLO_OFF, r, lh);
    }
#pragma unroll
    for (int t = 0; t < 2; ++t)
#pragma unroll
      for (int u = 0; u < 2; ++u) {
        // split-bf16 f32 emulation: hi*hi + hi*lo + lo*hi, f32 accumulate
        acc[t][u] = __builtin_amdgcn_wmma_f32_16x16x32_bf16(
            false, ah[t], false, bh[u], (short)0, acc[t][u], false, false);
        acc[t][u] = __builtin_amdgcn_wmma_f32_16x16x32_bf16(
            false, ah[t], false, bl[u], (short)0, acc[t][u], false, false);
        acc[t][u] = __builtin_amdgcn_wmma_f32_16x16x32_bf16(
            false, al[t], false, bh[u], (short)0, acc[t][u], false, false);
      }
  };

  const int nK = K >> 5;

  if (PRE) {
#if HAS_ASYNC_LDS
    stageAsync(0, lds);
    __builtin_amdgcn_s_wait_asynccnt(0);
    __syncthreads();
    for (int kb = 0; kb < nK; ++kb) {
      unsigned* cur = lds + (size_t)(kb & 1) * STAGE_U;
      const bool more = (kb + 1) < nK;
      if (more) stageAsync((kb + 1) << 5, lds + (size_t)((kb + 1) & 1) * STAGE_U);
      computeStep(cur);
      if (more) __builtin_amdgcn_s_wait_asynccnt(0);
      __syncthreads();
    }
#else
    loadStagePre(0);
    storeStagePre(lds);
    __syncthreads();
    for (int kb = 0; kb < nK; ++kb) {
      unsigned* cur = lds + (size_t)(kb & 1) * STAGE_U;
      const bool more = (kb + 1) < nK;
      if (more) loadStagePre((kb + 1) << 5);
      computeStep(cur);
      if (more) storeStagePre(lds + (size_t)((kb + 1) & 1) * STAGE_U);
      __syncthreads();
    }
#endif
  } else {
    loadStageF32(0);
    cvtStoreF32(lds);
    __syncthreads();
    for (int kb = 0; kb < nK; ++kb) {
      unsigned* cur = lds + (size_t)(kb & 1) * STAGE_U;
      const bool more = (kb + 1) < nK;
      if (more) loadStageF32((kb + 1) << 5);
      computeStep(cur);
      if (more) cvtStoreF32(lds + (size_t)((kb + 1) & 1) * STAGE_U);
      __syncthreads();
    }
  }

  // C/D layout: VGPR r, lanes 0-15 -> (M=r, N=lane); lanes 16-31 -> (M=r+8, N=lane-16)
#pragma unroll
  for (int t = 0; t < 2; ++t) {
#pragma unroll
    for (int u = 0; u < 2; ++u) {
      int col  = outBase + wRowBase + wn * 32 + u * 16 + ln;
      int row0 = rowBase + wm * 32 + t * 16 + lh * 8;
#pragma unroll
      for (int r = 0; r < 8; ++r) {
        out[(size_t)(row0 + r) * 4096 + col] = acc[t][u][r];
      }
    }
  }
}

extern "C" void kernel_launch(void* const* d_in, const int* in_sizes, int n_in,
                              void* d_out, int out_size, void* d_ws, size_t ws_size,
                              hipStream_t stream) {
  const float* x  = (const float*)d_in[0];
  const float* W0 = (const float*)d_in[1];
  const float* W1 = (const float*)d_in[2];
  const float* W2 = (const float*)d_in[3];
  float* out = (float*)d_out;

  const size_t X_ELEMS = 4ull * 4096 * 4096;                        // 67,108,864
  const size_t W_ELEMS = 1048576ull + 2097152ull + 8388608ull;      // 11,534,336
  const size_t need    = X_ELEMS * 4 + W_ELEMS * 4;                 // hi+lo bf16 (~300 MB)

  dim3 grid(32, 256);   // 32 N-blocks (8+8+16 across segments) x 256 M-blocks

  if (d_ws != nullptr && ws_size >= need) {
    unsigned* xhiU = (unsigned*)d_ws;                  // X_ELEMS/2 uints
    unsigned* xloU = xhiU + X_ELEMS / 2;
    unsigned* whiU = xloU + X_ELEMS / 2;               // W_ELEMS/2 uints
    unsigned* wloU = whiU + W_ELEMS / 2;

    convert_split_kernel<<<8192, 256, 0, stream>>>(x, xhiU, xloU, (int)(X_ELEMS / 4));
    convert_split_kernel<<<1024, 256, 0, stream>>>(W0, whiU,           wloU,           262144);
    convert_split_kernel<<<1024, 256, 0, stream>>>(W1, whiU + 524288,  wloU + 524288,  524288);
    convert_split_kernel<<<2048, 256, 0, stream>>>(W2, whiU + 1572864, wloU + 1572864, 2097152);

    matryoshka_wmma_kernel<true><<<grid, dim3(256), 0, stream>>>(
        x, W0, W1, W2, xhiU, xloU, whiU, wloU, out);
  } else {
    matryoshka_wmma_kernel<false><<<grid, dim3(256), 0, stream>>>(
        x, W0, W1, W2, nullptr, nullptr, nullptr, nullptr, out);
  }
}
